// BiLSTM_CRF_36498632081570
// MI455X (gfx1250) — matrix operand order repaired and checked by reading, not compiled, not run on hardware
//
#include <hip/hip_runtime.h>
#include <hip/hip_bf16.h>
#include <cstdint>
#include <cstddef>

// BiLSTM-CRF for MI455X (gfx1250, wave32, WMMA bf16->f32).
//
//  * All GEMMs via v_wmma_f32_16x16x32_bf16 with register double-buffered
//    fragment loads (software pipelining -> partial s_wait_loadcnt).
//  * Big time-parallel input projections Gx = x @ Wih^T: MT=2 x NT=4 register
//    blocking + global_prefetch_b8 two stages ahead (HBM-bound).
//  * Recurrent step GEMM (M=128, one launch per timestep in the graph):
//    MT=1 x NT=2 for wave parallelism; the Gx[t] tile is fetched by the
//    Tensor Data Mover (tensor_load_to_lds) at kernel entry, overlapping the
//    DMA with the whole K-loop; epilogue reads it from LDS after
//    s_wait_tensorcnt + barrier.  Falls back to global reads if the TDM
//    builtin is unavailable.
//  * Output projection (N=9) uses WMMA with an N-guard.
//  * CRF forward scan + backtrace in a single 128-thread block.

typedef __bf16 bf16;
typedef __attribute__((ext_vector_type(16))) __bf16 v16bf;
typedef __attribute__((ext_vector_type(8)))  float  v8f;
typedef __attribute__((ext_vector_type(4)))  unsigned int u32x4;
typedef __attribute__((ext_vector_type(4)))  int i32x4;
typedef __attribute__((ext_vector_type(8)))  int i32x8;

#if __has_builtin(__builtin_amdgcn_tensor_load_to_lds)
#define HAVE_TDM 1
#else
#define HAVE_TDM 0
#endif

#define DEV_ __device__ __forceinline__

namespace cfg {
constexpr int V = 30000, E = 512, HD = 1024, B = 128, T = 512, K = 9;
constexpr int H = 512, G4 = 2048;         // per-dir hidden, 4*H gate width
constexpr int START = 7, END = 8;
constexpr float NEG = -10000.0f;
}

union Frag { v16bf v; uint4 q[2]; };

DEV_ uint4 ld16(const bf16* p) { return *(const uint4*)p; }
DEV_ float sigm(float x) { return 1.0f / (1.0f + __expf(-x)); }

// ---------------------------------------------------------------------------
// Fragment loaders (layouts per CDNA5 ISA 7.12.2).
//   A 16x32 bf16: lane L -> row M=L%16, K = (L/16)*8 + {0..7}, 16+(L/16)*8+{0..7}
//   B 32x16 bf16: lane L -> col N=L%16, K = (L/16)*8 + {0..7}, 16+(L/16)*8+{0..7}
// arow[] already includes +halfk*8, wrow[] includes +halfk*8.
// ---------------------------------------------------------------------------
template <int MT, int NT, bool NGUARD>
DEV_ void load_frags(Frag a[MT], Frag b[NT],
                     const bf16* const arow[MT], const bf16* const wrow[NT],
                     int k0) {
#pragma unroll
    for (int i = 0; i < MT; ++i) {
        a[i].q[0] = ld16(arow[i] + k0);
        a[i].q[1] = ld16(arow[i] + k0 + 16);
    }
#pragma unroll
    for (int j = 0; j < NT; ++j) {
        if (!NGUARD || wrow[j]) {
            b[j].q[0] = ld16(wrow[j] + k0);
            b[j].q[1] = ld16(wrow[j] + k0 + 16);
        } else {
            b[j].q[0] = make_uint4(0u, 0u, 0u, 0u);
            b[j].q[1] = make_uint4(0u, 0u, 0u, 0u);
        }
    }
}

template <int MT, int NT>
DEV_ void mma_frags(const Frag a[MT], const Frag b[NT], v8f acc[MT][NT]) {
#pragma unroll
    for (int i = 0; i < MT; ++i)
#pragma unroll
        for (int j = 0; j < NT; ++j)
            acc[i][j] = __builtin_amdgcn_wmma_f32_16x16x32_bf16(
                false, a[i].v, false, b[j].v, (short)0, acc[i][j], false, false);
}

// ---------------------------------------------------------------------------
// Pipelined WMMA inner product: one wave computes a (16*MT) x (16*NT) tile of
// A(M,K) @ W(N,K)^T.  K must be a multiple of 64 (512 / 1024 here).
// ---------------------------------------------------------------------------
template <int MT, int NT, bool NGUARD, bool PF>
DEV_ void wmma_mk(const bf16* __restrict__ A, int lda,
                  const bf16* __restrict__ W, int ldw,
                  int mtile0, int ntile0, int N, int Kd, v8f acc[MT][NT]) {
    const int lane  = (int)(threadIdx.x & 31u);
    const int mloc  = lane & 15;
    const int halfk = lane >> 4;
    const bf16* arow[MT];
    const bf16* wrow[NT];
#pragma unroll
    for (int i = 0; i < MT; ++i)
        arow[i] = A + (size_t)((mtile0 + i) * 16 + mloc) * (size_t)lda + halfk * 8;
#pragma unroll
    for (int j = 0; j < NT; ++j) {
        const int n = (ntile0 + j) * 16 + mloc;
        wrow[j] = (!NGUARD || n < N)
                      ? (W + (size_t)n * (size_t)ldw + halfk * 8)
                      : nullptr;
    }

    Frag aP[MT], bP[NT], aQ[MT], bQ[NT];
    load_frags<MT, NT, NGUARD>(aP, bP, arow, wrow, 0);
    int k0 = 0;
    for (; k0 + 64 < Kd; k0 += 64) {
        load_frags<MT, NT, NGUARD>(aQ, bQ, arow, wrow, k0 + 32);
        if (PF) {   // speculative prefetch ~2 pipeline stages ahead
#pragma unroll
            for (int i = 0; i < MT; ++i) __builtin_prefetch(arow[i] + k0 + 128, 0, 1);
#pragma unroll
            for (int j = 0; j < NT; ++j)
                if (!NGUARD || wrow[j]) __builtin_prefetch(wrow[j] + k0 + 128, 0, 1);
        }
        mma_frags<MT, NT>(aP, bP, acc);
        load_frags<MT, NT, NGUARD>(aP, bP, arow, wrow, k0 + 64);
        mma_frags<MT, NT>(aQ, bQ, acc);
    }
    load_frags<MT, NT, NGUARD>(aQ, bQ, arow, wrow, k0 + 32);
    mma_frags<MT, NT>(aP, bP, acc);
    mma_frags<MT, NT>(aQ, bQ, acc);
}

// ---------------------------------------------------------------------------
// Generic GEMM: C(M,N) = A(M,K) @ W(N,K)^T (+ bias[n]).  4 waves / block.
// ---------------------------------------------------------------------------
template <int MT, int NT, bool OUT_BF16, bool NGUARD>
__global__ void gemm_bf16_kernel(const bf16* __restrict__ A,
                                 const bf16* __restrict__ W,
                                 const float* __restrict__ bias,
                                 void* __restrict__ Cout,
                                 int M, int N, int Kd, int ldc) {
    const int wave   = (int)(threadIdx.x >> 5);
    const int mtile0 = ((int)blockIdx.y * 4 + wave) * MT;
    const int ntile0 = (int)blockIdx.x * NT;
    v8f acc[MT][NT];
#pragma unroll
    for (int i = 0; i < MT; ++i)
#pragma unroll
        for (int j = 0; j < NT; ++j)
#pragma unroll
            for (int v = 0; v < 8; ++v) acc[i][j][v] = 0.0f;

    wmma_mk<MT, NT, NGUARD, true>(A, Kd, W, Kd, mtile0, ntile0, N, Kd, acc);

    const int lane  = (int)(threadIdx.x & 31u);
    const int mloc  = lane & 15;
    const int halfk = lane >> 4;
#pragma unroll
    for (int i = 0; i < MT; ++i)
#pragma unroll
        for (int j = 0; j < NT; ++j) {
            const int n = (ntile0 + j) * 16 + mloc;
            if (NGUARD && n >= N) continue;
            const float bv = bias ? bias[n] : 0.0f;
#pragma unroll
            for (int v = 0; v < 8; ++v) {
                const int m = (mtile0 + i) * 16 + halfk * 8 + v;
                const float val = acc[i][j][v] + bv;
                if (OUT_BF16) ((bf16*)Cout)[(size_t)m * ldc + n] = (bf16)val;
                else          ((float*)Cout)[(size_t)m * ldc + n] = val;
            }
        }
}

// ---------------------------------------------------------------------------
// Recurrent step GEMM: gates(dir,B,4H) = hprev(dir,B,H) @ Whh^T
//                      + biasSum(dir,4H) + Gx(dir,teff,B,4H)
// MT=1, NT=2; grid = (4H/32, B/64, 2 dirs), 128 threads, 4KB dynamic LDS.
// Gx tile (64 rows x 32 cols bf16) is DMA'd to LDS by the TDM while the
// K-loop runs; consumed after s_wait_tensorcnt + barrier.
// ---------------------------------------------------------------------------
__global__ void lstm_step_gemm_kernel(const bf16* __restrict__ hprev,
                                      const bf16* __restrict__ WhhF,
                                      const bf16* __restrict__ WhhB,
                                      const float* __restrict__ biasSum,
                                      const bf16* __restrict__ Gx,
                                      float* __restrict__ gates,
                                      int t, int T, int Bb, int Hh) {
    extern __shared__ bf16 gxTile[];                 // 64*32 bf16 = 4KB, lds_addr 0
    const int G4   = 4 * Hh;
    const int dir  = (int)blockIdx.z;
    const int teff = dir ? (T - 1 - t) : t;
    const bf16* A  = hprev + (size_t)dir * Bb * Hh;
    const bf16* W  = dir ? WhhB : WhhF;
    const bf16* gxBase = Gx + ((size_t)dir * T + teff) * (size_t)Bb * G4;
    const int m0 = (int)blockIdx.y * 64;             // block's batch-row base
    const int n0 = (int)blockIdx.x * 32;             // block's gate-col base

#if HAVE_TDM
    if ((threadIdx.x >> 5) == 0) {
        // Tensor DMA descriptor (CDNA5 ISA ch.8): 2-D tile, 2-byte elements,
        // tile_dim0=32 (contiguous cols), tile_dim1=64 (rows), row stride=G4.
        const uint64_t gaddr = (uint64_t)(uintptr_t)(gxBase + (size_t)m0 * G4 + n0);
        u32x4 g0;
        g0.x = 1u;                                          // count=1 (valid D#)
        g0.y = 0u;                                          // lds_addr = 0 (dynamic LDS base)
        g0.z = (unsigned)(gaddr & 0xFFFFFFFFu);             // global_addr[31:0]
        g0.w = (unsigned)((gaddr >> 32) & 0x01FFFFFFu)      // global_addr[56:32]
             | (2u << 30);                                  // type=2 ("image")
        i32x8 g1;
        g1[0] = (1 << 16);                                  // data_size=1 -> 2 bytes
        g1[1] = (int)((unsigned)(G4 & 0xFFFF) << 16);       // tensor_dim0[15:0]
        g1[2] = (int)(((unsigned)G4 >> 16) |
                      ((unsigned)(Bb & 0xFFFF) << 16));     // dim0[31:16] | dim1[15:0]
        g1[3] = (int)(((unsigned)Bb >> 16) | (32u << 16));  // dim1[31:16] | tile_dim0=32
        g1[4] = 64;                                         // tile_dim1=64, tile_dim2=0
        g1[5] = G4;                                         // tensor_dim0_stride[31:0]
        g1[6] = 0;                                          // stride hi | dim1_stride lo
        g1[7] = 0;
        i32x4 gz = {0, 0, 0, 0};
#if defined(__clang_major__) && (__clang_major__ >= 23)
        i32x8 gz8 = {0, 0, 0, 0, 0, 0, 0, 0};
        __builtin_amdgcn_tensor_load_to_lds(g0, g1, gz, gz, gz8, 0);
#else
        __builtin_amdgcn_tensor_load_to_lds(g0, g1, gz, gz, 0);
#endif
    }
#endif

    const int wave   = (int)(threadIdx.x >> 5);
    const int mtile0 = (int)blockIdx.y * 4 + wave;
    const int ntile0 = (int)blockIdx.x * 2;
    v8f acc[1][2];
#pragma unroll
    for (int j = 0; j < 2; ++j)
#pragma unroll
        for (int v = 0; v < 8; ++v) acc[0][j][v] = 0.0f;

    wmma_mk<1, 2, false, false>(A, Hh, W, Hh, mtile0, ntile0, G4, Hh, acc);

#if HAVE_TDM
    if ((threadIdx.x >> 5) == 0) {
#if __has_builtin(__builtin_amdgcn_s_wait_tensorcnt)
        __builtin_amdgcn_s_wait_tensorcnt(0);
#endif
    }
    __syncthreads();
#endif

    const int lane  = (int)(threadIdx.x & 31u);
    const int mloc  = lane & 15;
    const int halfk = lane >> 4;
    float* gout     = gates + (size_t)dir * Bb * G4;
#pragma unroll
    for (int j = 0; j < 2; ++j) {
        const int n = (ntile0 + j) * 16 + mloc;
        const float bv = biasSum[(size_t)dir * G4 + n];
#pragma unroll
        for (int v = 0; v < 8; ++v) {
            const int m = mtile0 * 16 + halfk * 8 + v;   // global batch row
#if HAVE_TDM
            const int lrow = m - m0;                      // wave*16 + halfk*8 + v
            const int lcol = j * 16 + mloc;
            const float gxv = (float)gxTile[lrow * 32 + lcol];
#else
            const float gxv = (float)gxBase[(size_t)m * G4 + n];
#endif
            gout[(size_t)m * G4 + n] = acc[0][j][v] + bv + gxv;
        }
    }
    (void)n0;
}

// ---------------------------------------------------------------------------
// Pointwise LSTM gates (order i,f,g,o). Writes c, hprev(bf16) and the
// per-layer output h buffer (T,B,HD) with dir-half offset.
// ---------------------------------------------------------------------------
__global__ void lstm_pointwise_kernel(const float* __restrict__ gates,
                                      float* __restrict__ cst,
                                      bf16* __restrict__ hprev,
                                      bf16* __restrict__ hout,
                                      int t, int T, int Bb, int Hh, int HDv) {
    const int dir = (int)blockIdx.z;
    const int idx = (int)(blockIdx.x * blockDim.x + threadIdx.x);
    const int total = Bb * Hh;
    if (idx >= total) return;
    const int b = idx / Hh, h = idx % Hh;
    const float* g = gates + ((size_t)dir * Bb + b) * (size_t)(4 * Hh);
    const float gi = g[h], gf = g[Hh + h], gg = g[2 * Hh + h], go = g[3 * Hh + h];
    const size_t ci = (size_t)dir * total + idx;
    const float c  = sigm(gf) * cst[ci] + sigm(gi) * tanhf(gg);
    const float hn = sigm(go) * tanhf(c);
    cst[ci]   = c;
    hprev[ci] = (bf16)hn;
    const int teff = dir ? (T - 1 - t) : t;
    hout[((size_t)teff * Bb + b) * HDv + dir * Hh + h] = (bf16)hn;
}

// ---------------------------------------------------------------------------
__global__ void embed_gather_kernel(const int* __restrict__ words,
                                    const float* __restrict__ emb,
                                    bf16* __restrict__ x0,
                                    int Bb, int T, int E) {
    const size_t total = (size_t)T * Bb * E;
    for (size_t i = (size_t)blockIdx.x * blockDim.x + threadIdx.x; i < total;
         i += (size_t)gridDim.x * blockDim.x) {
        const int e = (int)(i % E);
        const size_t r = i / E;
        const int b = (int)(r % Bb);
        const int t = (int)(r / Bb);
        const int w = words[(size_t)b * T + t];
        x0[i] = (bf16)emb[(size_t)w * E + e];
    }
}

__global__ void f32_to_bf16_kernel(const float* __restrict__ in,
                                   bf16* __restrict__ out, size_t n) {
    for (size_t i = (size_t)blockIdx.x * blockDim.x + threadIdx.x; i < n;
         i += (size_t)gridDim.x * blockDim.x)
        out[i] = (bf16)in[i];
}

__global__ void vecadd_kernel(const float* __restrict__ a,
                              const float* __restrict__ b,
                              float* __restrict__ out, int n) {
    const int i = (int)(blockIdx.x * blockDim.x + threadIdx.x);
    if (i < n) out[i] = a[i] + b[i];
}

// ---------------------------------------------------------------------------
// CRF: forward scan (logsumexp + argmax backpointers), final score mean,
// Viterbi backtrace. One block, thread b = batch row. K = 9.
// ---------------------------------------------------------------------------
__global__ void crf_kernel(const float* __restrict__ frames,
                           const float* __restrict__ trans,
                           int* __restrict__ bps,
                           float* __restrict__ out, int T, int Bb) {
    constexpr int K = cfg::K;
    __shared__ float tr[K * K];
    __shared__ float red[cfg::B];
    if (threadIdx.x < K * K) tr[threadIdx.x] = trans[threadIdx.x];
    __syncthreads();
    const int b = (int)threadIdx.x;

    float alpha[K];
#pragma unroll
    for (int k = 0; k < K; ++k) alpha[k] = (k == cfg::START) ? 0.0f : cfg::NEG;

    for (int t = 0; t < T; ++t) {
        const float* fr = frames + ((size_t)t * Bb + b) * K;
        float na[K];
#pragma unroll
        for (int cur = 0; cur < K; ++cur) {
            float m = -3.4e38f; int am = 0;
#pragma unroll
            for (int pv = 0; pv < K; ++pv) {
                const float s = alpha[pv] + tr[pv * K + cur];
                if (s > m) { m = s; am = pv; }
            }
            float sum = 0.0f;
#pragma unroll
            for (int pv = 0; pv < K; ++pv)
                sum += __expf(alpha[pv] + tr[pv * K + cur] - m);
            na[cur] = m + __logf(sum) + fr[cur];
            bps[((size_t)t * Bb + b) * K + cur] = am;
        }
#pragma unroll
        for (int cur = 0; cur < K; ++cur) alpha[cur] = na[cur];
    }

    float fin[K];
    float m = -3.4e38f; int best = 0;
#pragma unroll
    for (int cur = 0; cur < K; ++cur) {
        fin[cur] = alpha[cur] + tr[cur * K + cfg::END];
        if (fin[cur] > m) { m = fin[cur]; best = cur; }
    }
    float sum = 0.0f;
#pragma unroll
    for (int cur = 0; cur < K; ++cur) sum += __expf(fin[cur] - m);
    red[b] = m + __logf(sum);
    __syncthreads();
    for (int s = cfg::B / 2; s > 0; s >>= 1) {
        if (b < s) red[b] += red[b + s];
        __syncthreads();
    }
    if (b == 0) out[0] = red[0] / (float)Bb;

    // paths[b][i] = tag_{T-1-i}
    int tag = best;
    out[1 + (size_t)b * T + 0] = (float)tag;
    for (int i = 1; i < T; ++i) {
        const int t = T - i;                       // tag_{t-1} = bps[t][tag_t]
        tag = bps[((size_t)t * Bb + b) * K + tag];
        out[1 + (size_t)b * T + i] = (float)tag;
    }
}

// ---------------------------------------------------------------------------
extern "C" void kernel_launch(void* const* d_in, const int* in_sizes, int n_in,
                              void* d_out, int out_size, void* d_ws, size_t ws_size,
                              hipStream_t stream) {
    (void)in_sizes; (void)n_in; (void)out_size; (void)ws_size;
    using namespace cfg;
    const size_t TB = (size_t)T * B;   // 65536

    const int*   words = (const int*)  d_in[0];
    const float* emb   = (const float*)d_in[1];
    const float* Wih0f = (const float*)d_in[2],  *Whh0f = (const float*)d_in[3];
    const float* bih0f = (const float*)d_in[4],  *bhh0f = (const float*)d_in[5];
    const float* Wih0b = (const float*)d_in[6],  *Whh0b = (const float*)d_in[7];
    const float* bih0b = (const float*)d_in[8],  *bhh0b = (const float*)d_in[9];
    const float* Wih1f = (const float*)d_in[10], *Whh1f = (const float*)d_in[11];
    const float* bih1f = (const float*)d_in[12], *bhh1f = (const float*)d_in[13];
    const float* Wih1b = (const float*)d_in[14], *Whh1b = (const float*)d_in[15];
    const float* bih1b = (const float*)d_in[16], *bhh1b = (const float*)d_in[17];
    const float* Wout  = (const float*)d_in[18], *bout  = (const float*)d_in[19];
    const float* trans = (const float*)d_in[20];

    // ---- workspace carve -------------------------------------------------
    char* base = (char*)d_ws; size_t off = 0;
    auto carve = [&](size_t bytes) -> void* {
        void* r = base + off;
        off += (bytes + 255) & ~(size_t)255;
        return r;
    };
    bf16* Wih0f_b = (bf16*)carve((size_t)G4 * E * 2);
    bf16* Wih0b_b = (bf16*)carve((size_t)G4 * E * 2);
    bf16* Whh0f_b = (bf16*)carve((size_t)G4 * H * 2);
    bf16* Whh0b_b = (bf16*)carve((size_t)G4 * H * 2);
    bf16* Wih1f_b = (bf16*)carve((size_t)G4 * HD * 2);
    bf16* Wih1b_b = (bf16*)carve((size_t)G4 * HD * 2);
    bf16* Whh1f_b = (bf16*)carve((size_t)G4 * H * 2);
    bf16* Whh1b_b = (bf16*)carve((size_t)G4 * H * 2);
    bf16* Wout_b  = (bf16*)carve((size_t)K * HD * 2);
    float* biasL0 = (float*)carve((size_t)2 * G4 * 4);    // (dir,4H)
    float* biasL1 = (float*)carve((size_t)2 * G4 * 4);
    bf16* x0      = (bf16*)carve(TB * E * 2);             //  64 MB
    bf16* h0      = (bf16*)carve(TB * HD * 2);            // 128 MB
    bf16* h1      = (bf16*)carve(TB * HD * 2);            // 128 MB
    bf16* Gx      = (bf16*)carve((size_t)2 * TB * G4 * 2);// 512 MB (dir,T,B,4H)
    float* gates  = (float*)carve((size_t)2 * B * G4 * 4);
    float* cstate = (float*)carve((size_t)2 * B * H * 4);
    bf16* hprev   = (bf16*)carve((size_t)2 * B * H * 2);
    float* frames = (float*)carve(TB * K * 4);
    int*  bps     = (int*)carve(TB * K * 4);

    bf16* GxF = Gx;
    bf16* GxB = Gx + TB * G4;

    // ---- weight / bias prep ---------------------------------------------
    auto cvt = [&](const float* s, bf16* d, size_t n) {
        f32_to_bf16_kernel<<<dim3(1024), dim3(256), 0, stream>>>(s, d, n);
    };
    cvt(Wih0f, Wih0f_b, (size_t)G4 * E);
    cvt(Wih0b, Wih0b_b, (size_t)G4 * E);
    cvt(Whh0f, Whh0f_b, (size_t)G4 * H);
    cvt(Whh0b, Whh0b_b, (size_t)G4 * H);
    cvt(Wih1f, Wih1f_b, (size_t)G4 * HD);
    cvt(Wih1b, Wih1b_b, (size_t)G4 * HD);
    cvt(Whh1f, Whh1f_b, (size_t)G4 * H);
    cvt(Whh1b, Whh1b_b, (size_t)G4 * H);
    cvt(Wout,  Wout_b,  (size_t)K * HD);
    vecadd_kernel<<<dim3(G4 / 256), dim3(256), 0, stream>>>(bih0f, bhh0f, biasL0,       G4);
    vecadd_kernel<<<dim3(G4 / 256), dim3(256), 0, stream>>>(bih0b, bhh0b, biasL0 + G4,  G4);
    vecadd_kernel<<<dim3(G4 / 256), dim3(256), 0, stream>>>(bih1f, bhh1f, biasL1,       G4);
    vecadd_kernel<<<dim3(G4 / 256), dim3(256), 0, stream>>>(bih1b, bhh1b, biasL1 + G4,  G4);

    // ---- embedding gather ------------------------------------------------
    embed_gather_kernel<<<dim3(4096), dim3(256), 0, stream>>>(words, emb, x0, B, T, E);

    const dim3 gemmBlk(128);
    // MT=2,NT=4: block covers 128 rows x 64 cols
    const dim3 gxGrid(G4 / 64, (unsigned)(TB / 128));    // (32, 512)

    // ======================= layer 0 =======================
    gemm_bf16_kernel<2, 4, true, false><<<gxGrid, gemmBlk, 0, stream>>>(
        x0, Wih0f_b, nullptr, GxF, (int)TB, G4, E, G4);
    gemm_bf16_kernel<2, 4, true, false><<<gxGrid, gemmBlk, 0, stream>>>(
        x0, Wih0b_b, nullptr, GxB, (int)TB, G4, E, G4);

    hipMemsetAsync(cstate, 0, (size_t)2 * B * H * 4, stream);
    hipMemsetAsync(hprev,  0, (size_t)2 * B * H * 2, stream);

    const dim3 stepGrid(G4 / 32, B / 64, 2);             // (64, 2, 2)
    const dim3 pwGrid((B * H) / 256, 1, 2);
    const size_t stepLds = 64 * 32 * sizeof(bf16);       // 4 KB Gx tile
    for (int t = 0; t < T; ++t) {
        lstm_step_gemm_kernel<<<stepGrid, gemmBlk, stepLds, stream>>>(
            hprev, Whh0f_b, Whh0b_b, biasL0, Gx, gates, t, T, B, H);
        lstm_pointwise_kernel<<<pwGrid, dim3(256), 0, stream>>>(
            gates, cstate, hprev, h0, t, T, B, H, HD);
    }

    // ======================= layer 1 =======================
    gemm_bf16_kernel<2, 4, true, false><<<gxGrid, gemmBlk, 0, stream>>>(
        h0, Wih1f_b, nullptr, GxF, (int)TB, G4, HD, G4);
    gemm_bf16_kernel<2, 4, true, false><<<gxGrid, gemmBlk, 0, stream>>>(
        h0, Wih1b_b, nullptr, GxB, (int)TB, G4, HD, G4);

    hipMemsetAsync(cstate, 0, (size_t)2 * B * H * 4, stream);
    hipMemsetAsync(hprev,  0, (size_t)2 * B * H * 2, stream);

    for (int t = 0; t < T; ++t) {
        lstm_step_gemm_kernel<<<stepGrid, gemmBlk, stepLds, stream>>>(
            hprev, Whh1f_b, Whh1b_b, biasL1, Gx, gates, t, T, B, H);
        lstm_pointwise_kernel<<<pwGrid, dim3(256), 0, stream>>>(
            gates, cstate, hprev, h1, t, T, B, H, HD);
    }

    // ======================= emission projection (N=9) =======================
    gemm_bf16_kernel<1, 1, false, true><<<dim3(1, (unsigned)(TB / 64)), gemmBlk, 0, stream>>>(
        h1, Wout_b, bout, frames, (int)TB, K, HD, K);

    // ======================= CRF =======================
    crf_kernel<<<dim3(1), dim3(B), 0, stream>>>(frames, trans, bps, (float*)d_out, T, B);
}
